// Gnn_57200374448604
// MI455X (gfx1250) — compile-verified, matrix operand order checked by reading, not run on hardware
//
#include <hip/hip_runtime.h>
#include <hip/hip_bf16.h>

#define NNODES 30000
#define NEDGES 300000
#define DFEAT  768
#define CIN    2304   /* 3*D */
#define C2     3072   /* 4*D */
#define NCLS   16
#define ETOT   (NEDGES + NNODES)

typedef __attribute__((ext_vector_type(16))) __bf16 v16bf;
typedef __attribute__((ext_vector_type(8)))  float  v8f;

// ---------------- device helpers ----------------

__device__ __forceinline__ unsigned short f2bf(float f) {
  unsigned u = __float_as_uint(f);
  u += 0x7fffu + ((u >> 16) & 1u);       // round-to-nearest-even
  return (unsigned short)(u >> 16);
}
// monotonic float <-> uint encoding for atomicMax over signed floats
__device__ __forceinline__ unsigned encf(float f) {
  unsigned u = __float_as_uint(f);
  return (u >> 31) ? ~u : (u | 0x80000000u);
}
__device__ __forceinline__ float decf(unsigned u) {
  return __uint_as_float((u >> 31) ? (u & 0x7fffffffu) : ~u);
}

// ---------------- elementwise / stats kernels ----------------

__global__ void zero_kernel(float* p, int n) {
  int i = blockIdx.x * blockDim.x + threadIdx.x;
  if (i < n) p[i] = 0.0f;
}

__global__ void bn_stats_kernel(const float* __restrict__ x, float* __restrict__ sum,
                                float* __restrict__ sq, int n, int C) {
  int c = blockIdx.x * blockDim.x + threadIdx.x;
  if (c >= C) return;
  float s = 0.f, q = 0.f;
  for (int r = blockIdx.y; r < n; r += gridDim.y) {
    float v = x[(size_t)r * C + c];
    s += v; q += v * v;
  }
  atomicAdd(&sum[c], s);
  atomicAdd(&sq[c], q);
}

__global__ void bn_finalize_kernel(float* mean_io, float* invstd_io, int n, int C) {
  int c = blockIdx.x * blockDim.x + threadIdx.x;
  if (c >= C) return;
  float m = mean_io[c] / (float)n;
  float v = invstd_io[c] / (float)n - m * m;
  mean_io[c] = m;
  invstd_io[c] = rsqrtf(v + 1e-5f);
}

__global__ void bn_apply_cast_kernel(const float* __restrict__ x,
                                     const float* __restrict__ mean,
                                     const float* __restrict__ invstd,
                                     const float* __restrict__ gamma,
                                     const float* __restrict__ beta,
                                     unsigned short* __restrict__ xnb, int total, int C) {
  int i = blockIdx.x * blockDim.x + threadIdx.x;
  if (i >= total) return;
  int c = i % C;
  float v = (x[i] - mean[c]) * invstd[c] * gamma[c] + beta[c];
  xnb[i] = f2bf(v);
}

__global__ void cast_f2b_kernel(const float* __restrict__ in, unsigned short* __restrict__ out, int n) {
  int i = blockIdx.x * blockDim.x + threadIdx.x;
  if (i < n) out[i] = f2bf(in[i]);
}

// ---------------- WMMA GEMM: C[M,Nc] = A[M,K](bf16, split) * B[K,Nc](bf16) ----------------
// A element k: k < ksplit -> A0[row*lda0 + k], else A1[row*lda1 + (k-ksplit)]
// block = 256 threads (8 waves), block tile 64(M) x 256(N), wave tile 32x64, K-step 32.
// 8 v_wmma_f32_16x16x32_bf16 per wave per K-step; next tile's global loads are
// software-pipelined to issue before the WMMAs so HBM/L2 latency hides under compute.

__global__ __launch_bounds__(256) void gemm_bf16_wmma(
    const unsigned short* __restrict__ A0, int lda0, int ksplit,
    const unsigned short* __restrict__ A1, int lda1,
    const unsigned short* __restrict__ B, float* __restrict__ C,
    int M, int K, int Nc) {
  __shared__ unsigned ldsA[64 * 16];    // [row][kpair]            4 KB
  __shared__ unsigned ldsB[16 * 256];   // [kpair][col] packed    16 KB

  const int tid  = threadIdx.x;
  const int lane = tid & 31;
  const int wave = tid >> 5;
  const int waveM = wave & 1;   // rows  waveM*32
  const int waveN = wave >> 1;  // cols  waveN*64  (0..3)
  const int mBase = blockIdx.y * 64;
  const int nBase = blockIdx.x * 256;

  v8f acc[2][4];
#pragma unroll
  for (int i = 0; i < 2; ++i)
#pragma unroll
    for (int j = 0; j < 4; ++j) acc[i][j] = (v8f){0,0,0,0,0,0,0,0};

  // A loader: thread -> (row 0..63, 4 kpairs)
  const int arow = tid >> 2;
  const int acg  = tid & 3;
  int arow_g = mBase + arow; if (arow_g >= M) arow_g = M - 1;
  // B loader: thread -> (kpair 0..15, 16 cols)
  const int bkp = tid >> 4;
  const int bcg = tid & 15;

  const int r  = lane & 15;
  const int kh = lane >> 4;

  uint4 av, b0a, b0b, b1a, b1b;
  auto load_tiles = [&](int kt) {
    const unsigned short* Ap; int ld, kc;
    if (kt < ksplit) { Ap = A0; ld = lda0; kc = kt; }
    else             { Ap = A1; ld = lda1; kc = kt - ksplit; }
    av = *(const uint4*)(Ap + (size_t)arow_g * ld + kc + acg * 8);
    const int krow = kt + 2 * bkp;
    const unsigned short* b0p = B + (size_t)krow * Nc + nBase + bcg * 16;
    b0a = *(const uint4*)(b0p);
    b0b = *(const uint4*)(b0p + 8);
    b1a = *(const uint4*)(b0p + Nc);
    b1b = *(const uint4*)(b0p + Nc + 8);
  };

  load_tiles(0);   // prologue prefetch

  for (int kt = 0; kt < K; kt += 32) {
    __syncthreads();   // prior iteration's LDS reads done
    *(uint4*)&ldsA[arow * 16 + acg * 4] = av;
    {
      unsigned* bo = &ldsB[bkp * 256 + bcg * 16];
      uint4 p;
      p.x = (b0a.x & 0xffffu) | (b1a.x << 16);
      p.y = (b0a.x >> 16)     | (b1a.x & 0xffff0000u);
      p.z = (b0a.y & 0xffffu) | (b1a.y << 16);
      p.w = (b0a.y >> 16)     | (b1a.y & 0xffff0000u);
      *(uint4*)(bo + 0) = p;
      p.x = (b0a.z & 0xffffu) | (b1a.z << 16);
      p.y = (b0a.z >> 16)     | (b1a.z & 0xffff0000u);
      p.z = (b0a.w & 0xffffu) | (b1a.w << 16);
      p.w = (b0a.w >> 16)     | (b1a.w & 0xffff0000u);
      *(uint4*)(bo + 4) = p;
      p.x = (b0b.x & 0xffffu) | (b1b.x << 16);
      p.y = (b0b.x >> 16)     | (b1b.x & 0xffff0000u);
      p.z = (b0b.y & 0xffffu) | (b1b.y << 16);
      p.w = (b0b.y >> 16)     | (b1b.y & 0xffff0000u);
      *(uint4*)(bo + 8) = p;
      p.x = (b0b.z & 0xffffu) | (b1b.z << 16);
      p.y = (b0b.z >> 16)     | (b1b.z & 0xffff0000u);
      p.z = (b0b.w & 0xffffu) | (b1b.w << 16);
      p.w = (b0b.w >> 16)     | (b1b.w & 0xffff0000u);
      *(uint4*)(bo + 12) = p;
    }
    __syncthreads();   // LDS tile ready

    // prefetch next K tile while this tile's WMMAs execute
    if (kt + 32 < K) load_tiles(kt + 32);

    // ---- fragments (16-bit A/B VGPR layout: lane%16 = row/col, lane/16 = K half,
    //      VGPR v packs K pair; v 0..3 -> kp = kh*4+v, v 4..7 -> kp = 8+kh*4+(v-4))
    union FB { unsigned u[8]; v16bf v; };
    FB fa[2], fb[4];
    {
      const unsigned* base = &ldsA[(waveM * 32 + r) * 16 + kh * 4];
      *(uint4*)&fa[0].u[0] = *(const uint4*)(base);
      *(uint4*)&fa[0].u[4] = *(const uint4*)(base + 8);
      const unsigned* base1 = base + 16 * 16;
      *(uint4*)&fa[1].u[0] = *(const uint4*)(base1);
      *(uint4*)&fa[1].u[4] = *(const uint4*)(base1 + 8);
    }
#pragma unroll
    for (int j = 0; j < 4; ++j) {
      const int c0 = waveN * 64 + j * 16 + r;
#pragma unroll
      for (int v = 0; v < 4; ++v) {
        fb[j].u[v]     = ldsB[(kh * 4 + v) * 256 + c0];
        fb[j].u[v + 4] = ldsB[(8 + kh * 4 + v) * 256 + c0];
      }
    }
#pragma unroll
    for (int i = 0; i < 2; ++i)
#pragma unroll
      for (int j = 0; j < 4; ++j)
        acc[i][j] = __builtin_amdgcn_wmma_f32_16x16x32_bf16(
            false, fa[i].v, false, fb[j].v, (short)0, acc[i][j], false, false);
  }

  // ---- epilogue. C/D layout: row = 8*(lane/16) + v, col = lane%16
  const int colBase = nBase + waveN * 64 + r;
  const int rowTop  = mBase + waveM * 32 + kh * 8;
  if (mBase + 64 <= M) {
    // fast path: whole block tile in range, incremental row pointers
    float* p0 = C + (size_t)rowTop * Nc + colBase;
    float* p1 = p0 + (size_t)16 * Nc;
#pragma unroll
    for (int v = 0; v < 8; ++v) {
#pragma unroll
      for (int j = 0; j < 4; ++j) {
        p0[j * 16] = acc[0][j][v];
        p1[j * 16] = acc[1][j][v];
      }
      p0 += Nc; p1 += Nc;
    }
  } else {
#pragma unroll
    for (int v = 0; v < 8; ++v) {
#pragma unroll
      for (int i = 0; i < 2; ++i) {
        int row = rowTop + i * 16 + v;
        if (row < M) {
          float* p = C + (size_t)row * Nc + colBase;
#pragma unroll
          for (int j = 0; j < 4; ++j) p[j * 16] = acc[i][j][v];
        }
      }
    }
  }
}

// ---------------- attention kernels ----------------

__global__ void attn_dot_kernel(const float* __restrict__ h,
                                const float* __restrict__ as, const float* __restrict__ ad,
                                float* __restrict__ a_src, float* __restrict__ a_dst, int n) {
  int gid = blockIdx.x * blockDim.x + threadIdx.x;
  int node = gid >> 5, lane = gid & 31;
  if (node >= n) return;
  float s = 0.f, t = 0.f;
  for (int c = lane; c < DFEAT; c += 32) {
    float hv = h[(size_t)node * DFEAT + c];
    s += hv * as[c];
    t += hv * ad[c];
  }
  for (int o = 16; o > 0; o >>= 1) {
    s += __shfl_xor(s, o, 32);
    t += __shfl_xor(t, o, 32);
  }
  if (lane == 0) { a_src[node] = s; a_dst[node] = t; }
}

__global__ void init_node_kernel(unsigned* __restrict__ segmax, float* __restrict__ denom, int n) {
  int i = blockIdx.x * blockDim.x + threadIdx.x;
  if (i < n) { segmax[i] = encf(-__builtin_huge_valf()); denom[i] = 0.f; }
}

__global__ void edge_logit_kernel(const int* __restrict__ src, const int* __restrict__ dst,
                                  const float* __restrict__ a_src, const float* __restrict__ a_dst,
                                  float* __restrict__ logits, unsigned* __restrict__ segmax,
                                  int etot, int ebase) {
  int i = blockIdx.x * blockDim.x + threadIdx.x;
  if (i >= etot) return;
  int s, d;
  if (i < ebase) { s = src[i]; d = dst[i]; } else { s = d = i - ebase; }
  float z = a_src[s] + a_dst[d];
  z = (z > 0.f) ? z : 0.2f * z;           // leaky_relu(0.2)
  logits[i] = z;
  atomicMax(&segmax[d], encf(z));
}

__global__ void edge_exp_kernel(const int* __restrict__ dst,
                                const float* __restrict__ logits, const unsigned* __restrict__ segmax,
                                float* __restrict__ wexp, float* __restrict__ denom,
                                int etot, int ebase) {
  int i = blockIdx.x * blockDim.x + threadIdx.x;
  if (i >= etot) return;
  int d = (i < ebase) ? dst[i] : (i - ebase);
  float w = expf(logits[i] - decf(segmax[d]));
  wexp[i] = w;
  atomicAdd(&denom[d], w);
}

__global__ void init_out_kernel(float* __restrict__ out, const float* __restrict__ bias, int total) {
  int i = blockIdx.x * blockDim.x + threadIdx.x;
  if (i < total) out[i] = bias[i % DFEAT];
}

// 192 threads per edge, each handles 4 consecutive channels (float4 gather + 4 f32 atomics)
__global__ void scatter_kernel(const float* __restrict__ h,
                               const int* __restrict__ src, const int* __restrict__ dst,
                               const float* __restrict__ wexp, const float* __restrict__ denom,
                               float* __restrict__ out, int etot, int ebase) {
  int i = blockIdx.x * blockDim.x + threadIdx.x;
  int work = etot * (DFEAT / 4);
  if (i >= work) return;
  int e  = i / (DFEAT / 4);
  int c4 = (i % (DFEAT / 4)) * 4;
  int s, d;
  if (e < ebase) { s = src[e]; d = dst[e]; } else { s = d = e - ebase; }
  float alpha = wexp[e] / (denom[d] + 1e-16f);
  float4 hv = *(const float4*)(h + (size_t)s * DFEAT + c4);
  float* o = out + (size_t)d * DFEAT + c4;
  atomicAdd(o + 0, hv.x * alpha);
  atomicAdd(o + 1, hv.y * alpha);
  atomicAdd(o + 2, hv.z * alpha);
  atomicAdd(o + 3, hv.w * alpha);
}

// ---------------- head: pooler + loss ----------------

__global__ void pooler_kernel(const float* __restrict__ out2, const float* __restrict__ pW,
                              const float* __restrict__ pb, float* __restrict__ pool_out, int n) {
  int i = blockIdx.x * blockDim.x + threadIdx.x;
  if (i >= n * NCLS) return;
  int node = i >> 4, f = i & 15;
  float s = pb[f];
  const float* row = out2 + (size_t)node * DFEAT;
  for (int k = 0; k < DFEAT; ++k) s += row[k] * pW[k * NCLS + f];
  pool_out[i] = s;
}

__global__ void loss_kernel(const float* __restrict__ pooler, const int* __restrict__ target,
                            float* __restrict__ red, int n) {
  int i = blockIdx.x * blockDim.x + threadIdx.x;
  float ce = 0.f, ac = 0.f, cnt = 0.f;
  if (i < n) {
    int t = target[i];
    if (t >= 0) {
      const float* p = pooler + (size_t)i * NCLS;
      float m = p[0]; int am = 0;
      for (int j = 1; j < NCLS; ++j) if (p[j] > m) { m = p[j]; am = j; }
      float s = 0.f;
      for (int j = 0; j < NCLS; ++j) s += expf(p[j] - m);
      float lse = logf(s) + m;
      ce = lse - p[t];
      ac = (am == t) ? 1.f : 0.f;
      cnt = 1.f;
    }
  }
  __shared__ float sc[3];
  if (threadIdx.x == 0) { sc[0] = sc[1] = sc[2] = 0.f; }
  __syncthreads();
  atomicAdd(&sc[0], ce);
  atomicAdd(&sc[1], ac);
  atomicAdd(&sc[2], cnt);
  __syncthreads();
  if (threadIdx.x == 0) {
    atomicAdd(&red[0], sc[0]);
    atomicAdd(&red[1], sc[1]);
    atomicAdd(&red[2], sc[2]);
  }
}

__global__ void finalize_kernel(const float* __restrict__ red, float* __restrict__ scal) {
  scal[0] = red[0] / red[2];   // loss
  scal[1] = red[1] / red[2];   // acc
}

// ---------------- host ----------------

static inline int cdiv_ll(long long a, long long b) { return (int)((a + b - 1) / b); }

static void run_gat_attention(const float* h, const int* src, const int* dst,
                              const float* atts, const float* attd, const float* bias,
                              float* a_src, float* a_dst, unsigned* segmax, float* denom,
                              float* logits, float* wexp, float* out, hipStream_t stream) {
  attn_dot_kernel<<<cdiv_ll((long long)NNODES * 32, 256), 256, 0, stream>>>(h, atts, attd, a_src, a_dst, NNODES);
  init_node_kernel<<<cdiv_ll(NNODES, 256), 256, 0, stream>>>(segmax, denom, NNODES);
  edge_logit_kernel<<<cdiv_ll(ETOT, 256), 256, 0, stream>>>(src, dst, a_src, a_dst, logits, segmax, ETOT, NEDGES);
  edge_exp_kernel<<<cdiv_ll(ETOT, 256), 256, 0, stream>>>(dst, logits, segmax, wexp, denom, ETOT, NEDGES);
  init_out_kernel<<<cdiv_ll((long long)NNODES * DFEAT, 256), 256, 0, stream>>>(out, bias, NNODES * DFEAT);
  scatter_kernel<<<cdiv_ll((long long)ETOT * (DFEAT / 4), 256), 256, 0, stream>>>(h, src, dst, wexp, denom, out, ETOT, NEDGES);
}

extern "C" void kernel_launch(void* const* d_in, const int* in_sizes, int n_in,
                              void* d_out, int out_size, void* d_ws, size_t ws_size,
                              hipStream_t stream) {
  const float* x        = (const float*)d_in[0];
  const int*   ei       = (const int*)d_in[1];
  const int*   target   = (const int*)d_in[2];
  const float* bn_gamma = (const float*)d_in[3];
  const float* bn_beta  = (const float*)d_in[4];
  const float* W1       = (const float*)d_in[5];
  const float* att_src1 = (const float*)d_in[6];
  const float* att_dst1 = (const float*)d_in[7];
  const float* b1       = (const float*)d_in[8];
  const float* W2       = (const float*)d_in[9];
  const float* att_src2 = (const float*)d_in[10];
  const float* att_dst2 = (const float*)d_in[11];
  const float* b2       = (const float*)d_in[12];
  const float* pool_W   = (const float*)d_in[13];
  const float* pool_b   = (const float*)d_in[14];
  const int* src = ei;
  const int* dst = ei + NEDGES;

  float* out2   = (float*)d_out;
  float* pooler = out2 + (size_t)NNODES * DFEAT;
  float* scal   = pooler + (size_t)NNODES * NCLS;   // [loss, acc]

  char* ws = (char*)d_ws;
  size_t off = 0;
  auto alloc = [&](size_t bytes) -> void* {
    void* p = ws + off;
    off += (bytes + 255) & ~(size_t)255;
    return p;
  };
  float*          mean   = (float*)alloc((size_t)CIN * 4);
  float*          invstd = (float*)alloc((size_t)CIN * 4);
  unsigned short* xnb    = (unsigned short*)alloc((size_t)NNODES * CIN * 2);
  unsigned short* w1b    = (unsigned short*)alloc((size_t)CIN * DFEAT * 2);
  unsigned short* w2b    = (unsigned short*)alloc((size_t)C2 * DFEAT * 2);
  float*          h      = (float*)alloc((size_t)NNODES * DFEAT * 4);
  float*          out1   = (float*)alloc((size_t)NNODES * DFEAT * 4);
  unsigned short* out1b  = (unsigned short*)alloc((size_t)NNODES * DFEAT * 2);
  float*          a_src  = (float*)alloc((size_t)NNODES * 4);
  float*          a_dst  = (float*)alloc((size_t)NNODES * 4);
  unsigned*       segmax = (unsigned*)alloc((size_t)NNODES * 4);
  float*          denom  = (float*)alloc((size_t)NNODES * 4);
  float*          logits = (float*)alloc((size_t)ETOT * 4);
  float*          wexp   = (float*)alloc((size_t)ETOT * 4);
  float*          red    = (float*)alloc(16);
  (void)in_sizes; (void)n_in; (void)out_size; (void)ws_size;

  // ---- BatchNorm (training-mode batch stats) + cast to bf16
  zero_kernel<<<cdiv_ll(CIN, 256), 256, 0, stream>>>(mean, CIN);
  zero_kernel<<<cdiv_ll(CIN, 256), 256, 0, stream>>>(invstd, CIN);
  bn_stats_kernel<<<dim3(CIN / 256, 64), 256, 0, stream>>>(x, mean, invstd, NNODES, CIN);
  bn_finalize_kernel<<<cdiv_ll(CIN, 256), 256, 0, stream>>>(mean, invstd, NNODES, CIN);
  bn_apply_cast_kernel<<<cdiv_ll((long long)NNODES * CIN, 256), 256, 0, stream>>>(
      x, mean, invstd, bn_gamma, bn_beta, xnb, NNODES * CIN, CIN);

  // ---- weight casts to bf16
  cast_f2b_kernel<<<cdiv_ll((long long)CIN * DFEAT, 256), 256, 0, stream>>>(W1, w1b, CIN * DFEAT);
  cast_f2b_kernel<<<cdiv_ll((long long)C2 * DFEAT, 256), 256, 0, stream>>>(W2, w2b, C2 * DFEAT);

  dim3 gemm_grid(DFEAT / 256, cdiv_ll(NNODES, 64));

  // ---- GAT layer 1: h = xn @ W1, attention, scatter -> out1
  gemm_bf16_wmma<<<gemm_grid, 256, 0, stream>>>(xnb, CIN, CIN, xnb, CIN, w1b, h, NNODES, CIN, DFEAT);
  run_gat_attention(h, src, dst, att_src1, att_dst1, b1,
                    a_src, a_dst, segmax, denom, logits, wexp, out1, stream);

  // ---- GAT layer 2: A = [out1 | xn] (split-K GEMM), attention -> out2 (d_out)
  cast_f2b_kernel<<<cdiv_ll((long long)NNODES * DFEAT, 256), 256, 0, stream>>>(out1, out1b, NNODES * DFEAT);
  gemm_bf16_wmma<<<gemm_grid, 256, 0, stream>>>(out1b, DFEAT, DFEAT, xnb, CIN, w2b, h, NNODES, C2, DFEAT);
  run_gat_attention(h, src, dst, att_src2, att_dst2, b2,
                    a_src, a_dst, segmax, denom, logits, wexp, out2, stream);

  // ---- pooler + loss/acc
  pooler_kernel<<<cdiv_ll((long long)NNODES * NCLS, 256), 256, 0, stream>>>(out2, pool_W, pool_b, pooler, NNODES);
  zero_kernel<<<1, 4, 0, stream>>>(red, 3);
  loss_kernel<<<cdiv_ll(NNODES, 256), 256, 0, stream>>>(pooler, target, red, NNODES);
  finalize_kernel<<<1, 1, 0, stream>>>(red, scal);
}
